// _ReduceBase_58110907515667
// MI455X (gfx1250) — matrix operand order repaired and checked by reading, not compiled, not run on hardware
//
#include <hip/hip_runtime.h>
#include <hip/hip_bf16.h>
#include <math.h>

// ---------------------------------------------------------------------------
// Gated tree-reduction for CDNA5 (gfx1250):
//   V_WMMA_F32_16X16X32_F16 GEMM core (f16 in, f32 accumulate, K=1024)
//   + GLOBAL_LOAD_ASYNC_TO_LDS_B128 staging of the B operand (ASYNCcnt path)
// ---------------------------------------------------------------------------

typedef _Float16 v16h __attribute__((ext_vector_type(16)));
typedef float    v8f  __attribute__((ext_vector_type(8)));

union V16H {
    v16h        v;
    unsigned    u[8];
    _Float16    h[16];
};

__device__ __forceinline__ v8f wmma_f16(v16h a, v16h b, v8f c) {
    // (neg_a, A, neg_b, B, c_mod, C, reuse_a, reuse_b)
    return __builtin_amdgcn_wmma_f32_16x16x32_f16(false, a, false, b,
                                                  (short)0, c, false, false);
}

// ---------------------------------------------------------------------------
// One-time: convert W (f32, [1536][512][2] = [o][kappa], kappa = 2c+k) into
// a contiguous-k 16-bit A-matrix lane order (same k order as the B fill):
//   lane<16 -> M=lane,   halves 0..15 = K 0..15
//   lane>=16-> M=lane-16, halves 0..15 = K 16..31
// Wsw[((mt*32 + kc)*32 + lane)*16 + hh]  -> one 32B contiguous A load per lane.
// ---------------------------------------------------------------------------
__global__ void prep_w_kernel(const float* __restrict__ W,
                              _Float16* __restrict__ Wsw) {
    int id = blockIdx.x * 256 + threadIdx.x;           // 96*32*32*16 = 1572864
    if (id >= 96 * 32 * 32 * 16) return;
    int hh   = id & 15;
    int lane = (id >> 4) & 31;
    int kc   = (id >> 9) & 31;
    int mt   = id >> 14;
    int M    = mt * 16 + (lane & 15);
    int hi   = lane >> 4;
    int kl   = hi * 16 + hh;
    int kappa = kc * 32 + kl;
    Wsw[id] = (_Float16)W[(size_t)M * 1024 + kappa];
}

__device__ __forceinline__ float gate_combine(float cl, float cr, float cg,
                                              const float* __restrict__ bias,
                                              int o) {
    float lv = cl + bias[o];
    float rv = tanhf(cr + bias[512 + o]);
    float gx = cg + bias[1024 + o];
    float gv = 1.0f / (1.0f + expf(-gx));
    return lv * gv + rv * (1.0f - gv);
}

// Shared K-loop: B fragments come from the LDS slab sb[c][16] (dwords = f16
// pairs, kappa = 2c+k). Bank-conflict-free for the WMMA B lane pattern.
__device__ __forceinline__ void gemm_from_lds(const unsigned* sb,
                                              const v16h* __restrict__ Wsw,
                                              int m, int hi, int ln,
                                              int lane,
                                              v8f& cl, v8f& cr, v8f& cg) {
    for (int kc = 0; kc < 32; ++kc) {
        V16H B;
        int cbase = kc * 16 + hi * 8;
        #pragma unroll
        for (int v = 0; v < 8; ++v)
            B.u[v] = sb[(cbase + v) * 16 + ln];
        v16h a0 = Wsw[((size_t)m * 32 + kc) * 32 + lane];
        v16h a1 = Wsw[((size_t)(m + 32) * 32 + kc) * 32 + lane];
        v16h a2 = Wsw[((size_t)(m + 64) * 32 + kc) * 32 + lane];
        cl = wmma_f16(a0, B.v, cl);
        cr = wmma_f16(a1, B.v, cr);
        cg = wmma_f16(a2, B.v, cg);
    }
}

// ---------------------------------------------------------------------------
// Stage A: cur0[b,c,j] (width 4096, f16) =
//    j <  half : gated-reduce of f32 h pairs (2j, 2j+1)   [GEMM]
//    j >= half : h[b, c, min(j+half, 4095)]               [copy]
// The LDS fill does the f32->f16 conversion once per block (hoists cvt out of
// the WMMA loop).
// ---------------------------------------------------------------------------
__global__ void __launch_bounds__(128)
stageA_kernel(const float* __restrict__ h,
              const v16h* __restrict__ Wsw,
              const float* __restrict__ bias,
              const int* __restrict__ N,
              _Float16* __restrict__ out) {
    __shared__ unsigned sb[8192];                      // 32KB: [c:512][ln:16]
    const int b    = blockIdx.z;
    const int tid  = threadIdx.x;
    const int wave = tid >> 5;
    const int lane = tid & 31;
    const int m    = blockIdx.y * 4 + wave;            // 0..31 (l-section tile)
    const int hi   = lane >> 4;
    const int ln   = lane & 15;
    const int j0   = blockIdx.x * 16;
    const int jcol = j0 + ln;

    const int n    = N[b];
    const int nf   = 1 << (31 - __builtin_clz((unsigned)n));
    const int half = n - nf;                           // 0..2047

    const float* hb   = h   + (size_t)b * 512 * 4096;
    _Float16*    outb = out + (size_t)b * 512 * 4096;

    if (j0 >= half) {                                  // uniform per block
        #pragma unroll
        for (int v = 0; v < 8; ++v) {
            int c   = m * 16 + v + 8 * hi;
            int idx = min(jcol + half, 4095);
            outb[(size_t)c * 4096 + jcol] = (_Float16)hb[(size_t)c * 4096 + idx];
        }
        return;
    }

    // LDS fill with conversion: 8192 dwords, 64 per thread.
    for (int d = tid; d < 8192; d += 128) {
        int c   = d >> 4;
        int lnd = d & 15;
        int jl  = min(j0 + lnd, 2047);                 // pair reads in-bounds
        const float* q = hb + (size_t)c * 4096 + 2 * jl;
        union { _Float16 h2[2]; unsigned u; } p;
        p.h2[0] = (_Float16)q[0];
        p.h2[1] = (_Float16)q[1];
        sb[d] = p.u;
    }
    __syncthreads();

    v8f cl = {}, cr = {}, cg = {};
    gemm_from_lds(sb, Wsw, m, hi, ln, lane, cl, cr, cg);

    const bool take_red = jcol < half;
    #pragma unroll
    for (int v = 0; v < 8; ++v) {
        int   o   = m * 16 + v + 8 * hi;
        float res = gate_combine(cl[v], cr[v], cg[v], bias, o);
        float fin = take_red ? res
                             : hb[(size_t)o * 4096 + min(jcol + half, 4095)];
        outb[(size_t)o * 4096 + jcol] = (_Float16)fin;
    }
}

// ---------------------------------------------------------------------------
// Level t: read f16 cur (width w_in), write width cols = w_in/2.
// B slab (512 rows x 64B) staged into LDS once per block via
// GLOBAL_LOAD_ASYNC_TO_LDS_B128 (tracked by ASYNCcnt), then consumed by all
// 4 waves -> 4x less global B traffic than per-wave loads.
// ---------------------------------------------------------------------------
__global__ void __launch_bounds__(128)
level_kernel(const _Float16* __restrict__ in,
             const v16h* __restrict__ Wsw,
             const float* __restrict__ bias,
             const int* __restrict__ N,
             _Float16* __restrict__ out,
             int t, int w_in, int cols) {
    __shared__ unsigned sb[8192];                      // 32KB: [c:512][ln:16]
    const int b    = blockIdx.z;
    const int tid  = threadIdx.x;
    const int wave = tid >> 5;
    const int lane = tid & 31;
    const int m    = blockIdx.y * 4 + wave;
    const int hi   = lane >> 4;
    const int ln   = lane & 15;
    const int j0   = blockIdx.x * 16;
    const int jcol = j0 + ln;

    const int n  = N[b];
    const int nf = 1 << (31 - __builtin_clz((unsigned)n));
    const bool active = (nf >> t) > 1;                 // uniform per block

    const _Float16* inb  = in  + (size_t)b * 512 * w_in;
    _Float16*       outb = out + (size_t)b * 512 * cols;

    if (!active) {                                     // uniform: copy prefix
        #pragma unroll
        for (int v = 0; v < 8; ++v) {
            int c = m * 16 + v + 8 * hi;
            if (jcol < cols)
                outb[(size_t)c * cols + jcol] = inb[(size_t)c * w_in + jcol];
        }
        return;
    }

    const unsigned* pB = (const unsigned*)inb;         // 2 halves per dword
    const int halfw = w_in >> 1;                       // dwords per c-row

    if ((halfw & 3) == 0) {
        // Async staged copy: thread -> one b128 (16B) per round, 16 rounds.
        const int c0   = tid >> 2;                     // row for round 0
        const int dcol = (tid & 3) * 4;                // dword offset in row
        #pragma unroll
        for (int it = 0; it < 16; ++it) {
            int c = it * 32 + c0;
            unsigned lds_off =
                (unsigned)(unsigned long long)(size_t)&sb[(c << 4) + dcol];
            unsigned long long gaddr =
                (unsigned long long)(size_t)(pB + (size_t)c * halfw + j0 + dcol);
            asm volatile("global_load_async_to_lds_b128 %0, %1, off"
                         :: "v"(lds_off), "v"(gaddr) : "memory");
        }
        asm volatile("s_wait_asynccnt 0x0" ::: "memory");
    } else {
        // Tiny levels (w_in = 4, 2): simple dword staging, clamped columns.
        for (int d = tid; d < 8192; d += 128) {
            int c   = d >> 4;
            int lnd = d & 15;
            sb[d] = pB[(size_t)c * halfw + min(j0 + lnd, cols - 1)];
        }
    }
    __syncthreads();

    v8f cl = {}, cr = {}, cg = {};
    gemm_from_lds(sb, Wsw, m, hi, ln, lane, cl, cr, cg);

    #pragma unroll
    for (int v = 0; v < 8; ++v) {
        int   o   = m * 16 + v + 8 * hi;
        float res = gate_combine(cl[v], cr[v], cg[v], bias, o);
        if (jcol < cols)
            outb[(size_t)o * cols + jcol] = (_Float16)res;
    }
}

// ---------------------------------------------------------------------------
// Final: d_out[b*512 + c] = (float) cur_final[b, c, 0]  (width-1 buffer)
// ---------------------------------------------------------------------------
__global__ void finalize_kernel(const _Float16* __restrict__ q,
                                float* __restrict__ outp) {
    int i = blockIdx.x * 256 + threadIdx.x;
    if (i < 16 * 512) outp[i] = (float)q[i];
}

// ---------------------------------------------------------------------------
// Workspace layout (bytes):
//   R   @ 0         : 16*512*4096*2 = 67108864   (stage-A output, width 4096)
//   P   @ 67108864  : 16*512*2048*2 = 33554432   (ping, width <= 2048)
//   Q   @ 100663296 : 16*512*1024*2 = 16777216   (pong, width <= 1024)
//   Wsw @ 117440512 : 1536*1024*2   = 3145728    (swizzled f16 weights)
// ---------------------------------------------------------------------------
extern "C" void kernel_launch(void* const* d_in, const int* in_sizes, int n_in,
                              void* d_out, int out_size, void* d_ws, size_t ws_size,
                              hipStream_t stream) {
    const float* h    = (const float*)d_in[0];
    const float* W    = (const float*)d_in[1];
    const float* bias = (const float*)d_in[2];
    const int*   N    = (const int*)d_in[3];

    char* ws = (char*)d_ws;
    _Float16* R   = (_Float16*)(ws);
    _Float16* P   = (_Float16*)(ws + 67108864u);
    _Float16* Q   = (_Float16*)(ws + 67108864u + 33554432u);
    _Float16* Wsw = (_Float16*)(ws + 67108864u + 33554432u + 16777216u);

    // 1) swizzle weights to WMMA A-layout
    prep_w_kernel<<<6144, 256, 0, stream>>>(W, Wsw);

    // 2) stage A: initial reduce + suffix splice, width 4096 output
    stageA_kernel<<<dim3(256, 8, 16), 128, 0, stream>>>(
        h, (const v16h*)Wsw, bias, N, R);

    // 3) 12 tree-reduction levels, ping-pong P/Q
    const _Float16* src = R;
    _Float16* bufs[2] = {P, Q};
    int w_in = 4096;
    for (int t = 0; t < 12; ++t) {
        int cols = w_in >> 1;
        int gx   = (cols + 15) / 16;
        _Float16* dst = bufs[t & 1];
        level_kernel<<<dim3(gx, 8, 16), 128, 0, stream>>>(
            src, (const v16h*)Wsw, bias, N, dst, t, w_in, cols);
        src  = dst;
        w_in = cols;
    }

    // 4) emit (16, 512) f32
    finalize_kernel<<<32, 256, 0, stream>>>(src, (float*)d_out);
}
